// HybridGNN_RNN_13666585935966
// MI455X (gfx1250) — matrix operand order repaired and checked by reading, not compiled, not run on hardware
//
#include <hip/hip_runtime.h>
#include <hip/hip_bf16.h>
#include <math.h>

typedef __attribute__((ext_vector_type(16))) _Float16 v16h;
typedef __attribute__((ext_vector_type(8)))  _Float16 v8h;
typedef __attribute__((ext_vector_type(8)))  float    v8f;

#define N_NODES 8192
#define N_EDGES 262144
#define T_STEPS 32
#define D_DYN   16
#define D_DYNP  32      // K padded to WMMA depth
#define D_STAT  128
#define G_HID   64
#define R_HID   128
#define N_OUT   10
#define N_OUTP  16      // N padded to tile width
#define EPSV    1e-5f

// ---------------------------------------------------------------------------
// A fragment: f16 row-major (M, lda), K multiple of 32. Per ISA 16-bit A 16x32
// layout, lane (l%16)=M-row, g=l/16; v16h = halves [k0+8g..+7] ++ [k0+16+8g..+7]
// -> two 16B global_load_b128.
// ---------------------------------------------------------------------------
__device__ __forceinline__ v16h load_a_frag(const _Float16* __restrict__ A, int lda,
                                            int m, int k0) {
  int g = (threadIdx.x >> 4) & 1;
  const _Float16* p = A + (size_t)m * lda + k0 + 8 * g;
  v8h lo = *(const v8h*)p;          // 16B aligned: lda, k0+8g multiples of 8
  v8h hi = *(const v8h*)(p + 16);
  v16h a;
#pragma unroll
  for (int i = 0; i < 8; ++i) { a[i] = lo[i]; a[8 + i] = hi[i]; }
  return a;
}

// B fragment out of the LDS stage. LDS holds the block's B^T tile:
// (16*NSUB) rows x K halves, row-contiguous. Lane n = nloc+l%16, group g holds
// K = [k0+8g..+7] ++ [k0+16+8g..+7] -> two 16B ds_load_b128 from row nloc.
__device__ __forceinline__ v16h load_b_lds(const _Float16* __restrict__ lb, int K,
                                           int k0, int nloc) {
  int lane = threadIdx.x & 31;
  const _Float16* p = lb + (size_t)(nloc + (lane & 15)) * K + k0 + 8 * (lane >> 4);
  v8h lo = *(const v8h*)p;
  v8h hi = *(const v8h*)(p + 16);
  v16h b;
#pragma unroll
  for (int i = 0; i < 8; ++i) { b[i] = lo[i]; b[8 + i] = hi[i]; }
  return b;
}

// ---------------------------------------------------------------------------
// Generic WMMA GEMM: out = act(bn(A@B^T + Cin + bias1 + bias2))
// One block (4 waves) computes a 64 x (16*NSUB) macro-tile; the shared B^T
// tile is staged once in LDS, each wave owns 16 rows. Host guarantees:
// M % 64 == 0, K % 32 == 0, Npad % 16 == 0, ldb == K; store guarded by Nreal.
// ---------------------------------------------------------------------------
template <int NSUB>
__global__ void __launch_bounds__(128) wmma_gemm(
    const _Float16* __restrict__ A, int lda,
    const _Float16* __restrict__ Bt,
    const float* __restrict__ Cin, int ldcin,
    const float* __restrict__ bias1, const float* __restrict__ bias2,
    const float* __restrict__ bng, const float* __restrict__ bnb,
    int relu,
    float* __restrict__ C, _Float16* __restrict__ Ch, int ldc,
    int M, int Nreal, int Npad, int K) {
  extern __shared__ char smem_raw[];
  _Float16* lb = (_Float16*)smem_raw;              // (16*NSUB) x K halves

  const int lane = threadIdx.x & 31;
  const int tiles_n = Npad / (16 * NSUB);
  const int btm = (blockIdx.x / tiles_n) * 64;     // exact grid: no tail blocks
  const int tn0 = (blockIdx.x % tiles_n) * 16 * NSUB;

  // ---- stage B^T tile (16*NSUB rows x K) into LDS, 16B chunks, coalesced ----
  const int kv = K >> 3;                            // 16B chunks per row
  const int chunks = 16 * NSUB * kv;
  for (int idx = threadIdx.x; idx < chunks; idx += blockDim.x) {
    int row = idx / kv;
    int col = (idx - row * kv) << 3;
    *(v8h*)(lb + row * K + col) = *(const v8h*)(Bt + (size_t)(tn0 + row) * K + col);
  }
  __syncthreads();

  const int tm   = btm + 16 * (threadIdx.x >> 5);   // wave's 16-row slice
  const int mrow = tm + (lane & 15);

  v8f acc[NSUB] = {};
  v16h af = load_a_frag(A, lda, mrow, 0);
  for (int k0 = 0; k0 < K; k0 += 32) {
    v16h af_next;
    if (k0 + 32 < K) {
      __builtin_prefetch(A + (size_t)mrow * lda + k0 + 64, 0, 1);  // global_prefetch_b8
      af_next = load_a_frag(A, lda, mrow, k0 + 32);                // overlap w/ WMMAs
    }
#pragma unroll
    for (int s = 0; s < NSUB; ++s) {
      v16h bf = load_b_lds(lb, K, k0, 16 * s);
      acc[s] = __builtin_amdgcn_wmma_f32_16x16x32_f16(
          false, af, false, bf, (short)0, acc[s], false, false);
    }
    af = af_next;
  }

  const float bns = rsqrtf(1.0f + EPSV);
#pragma unroll
  for (int s = 0; s < NSUB; ++s) {
    int n = tn0 + 16 * s + (lane & 15);
    if (n >= Nreal) continue;   // only the 10-wide head pads N
#pragma unroll
    for (int r = 0; r < 8; ++r) {
      int m = tm + r + 8 * (lane >> 4);
      float v = acc[s][r];
      if (Cin)   v += Cin[(size_t)m * ldcin + n];
      if (bias1) v += bias1[n];
      if (bias2) v += bias2[n];
      if (bng)   v = v * (bng[n] * bns) + bnb[n];
      if (relu)  v = fmaxf(v, 0.0f);
      if (C)  C[(size_t)m * ldc + n] = v;
      if (Ch) Ch[(size_t)m * ldc + n] = (_Float16)v;
    }
  }
}

// ---------------------------------------------------------------------------
// Conversion kernels (fp32 -> f16 operand staging, run every call)
// ---------------------------------------------------------------------------
// dst (N, Kp) f16 <- src (N, K) f32, zero-pad k >= K  (plain cast if K==Kp)
__global__ void conv_padk_kernel(const float* __restrict__ src, _Float16* __restrict__ dst,
                                 int N, int K, int Kp) {
  int id = blockIdx.x * blockDim.x + threadIdx.x;
  if (id < N * Kp) {
    int n = id / Kp, k = id - n * Kp;
    dst[id] = (_Float16)((k < K) ? src[n * K + k] : 0.0f);
  }
}

// dst (Np, K) f16 <- transpose of src (K, N) f32, zero-pad n >= N
__global__ void conv_transpose_kernel(const float* __restrict__ src, _Float16* __restrict__ dst,
                                      int K, int N, int Np) {
  int id = blockIdx.x * blockDim.x + threadIdx.x;
  if (id < Np * K) {
    int n = id / K, k = id - n * K;
    dst[id] = (_Float16)((n < N) ? src[k * N + n] : 0.0f);
  }
}

// ---------------------------------------------------------------------------
// Elementwise / graph kernels
// ---------------------------------------------------------------------------
__global__ void fill_kernel(float* __restrict__ p, float v, int n) {
  int i = blockIdx.x * blockDim.x + threadIdx.x;
  if (i < n) p[i] = v;
}

__global__ void deg_accum_kernel(const int* __restrict__ dst, float* __restrict__ deg, int E) {
  int e = blockIdx.x * blockDim.x + threadIdx.x;
  if (e < E) atomicAdd(&deg[dst[e]], 1.0f);
}

__global__ void dis_kernel(const float* __restrict__ deg, float* __restrict__ dis, int n) {
  int i = blockIdx.x * blockDim.x + threadIdx.x;
  if (i < n) dis[i] = rsqrtf(fmaxf(deg[i], 1.0f));
}

// agg[dst] += h[src] * dis[src]*dis[dst]   (one thread per (edge, feature))
__global__ void gcn_scatter_kernel(const float* __restrict__ h, const int* __restrict__ src,
                                   const int* __restrict__ dst, const float* __restrict__ dis,
                                   float* __restrict__ agg, int E, int F) {
  int id = blockIdx.x * blockDim.x + threadIdx.x;
  int e = id / F, f = id - e * F;
  if (e < E) {
    int s = src[e], d = dst[e];
    atomicAdd(&agg[d * F + f], h[s * F + f] * dis[s] * dis[d]);
  }
}

// out = relu( bn( agg + h*dis^2 (self loop) + gcn_bias ) ); writes f32 and f16
__global__ void gcn_finish_kernel(const float* __restrict__ agg, const float* __restrict__ h,
                                  const float* __restrict__ dis, const float* __restrict__ gb,
                                  const float* __restrict__ bng, const float* __restrict__ bnb,
                                  float* __restrict__ out, _Float16* __restrict__ out_h,
                                  int Nn, int F) {
  int id = blockIdx.x * blockDim.x + threadIdx.x;
  int i = id / F, f = id - i * F;
  if (i < Nn) {
    float di = dis[i];
    float v = agg[id] + h[id] * di * di + gb[f];
    v = v * (bng[f] * rsqrtf(1.0f + EPSV)) + bnb[f];
    v = fmaxf(v, 0.0f);
    out[id] = v;
    out_h[id] = (_Float16)v;
  }
}

// LSTM cell: gates layout [i | f | g | o]; writes h (f32 + f16) and c in place
__global__ void lstm_cell_kernel(const float* __restrict__ gates, int ldg,
                                 float* __restrict__ h, _Float16* __restrict__ h_h,
                                 float* __restrict__ c, int M, int H) {
  int id = blockIdx.x * blockDim.x + threadIdx.x;
  int m = id / H, j = id - m * H;
  if (m < M) {
    const float* g = gates + (size_t)m * ldg;
    float ig = 1.0f / (1.0f + __expf(-g[j]));
    float fg = 1.0f / (1.0f + __expf(-g[H + j]));
    float gg = tanhf(g[2 * H + j]);
    float og = 1.0f / (1.0f + __expf(-g[3 * H + j]));
    float cc = fg * c[id] + ig * gg;
    float hh = og * tanhf(cc);
    c[id] = cc;
    h[id] = hh;
    h_h[id] = (_Float16)hh;
  }
}

// fused f16 = concat(gnn f32 [Fa], rnn f32 [Fb])
__global__ void concat_kernel(const float* __restrict__ a, int Fa,
                              const float* __restrict__ b, int Fb,
                              _Float16* __restrict__ out, int M) {
  int F = Fa + Fb;
  int id = blockIdx.x * blockDim.x + threadIdx.x;
  int m = id / F, f = id - m * F;
  if (m < M) out[id] = (_Float16)((f < Fa) ? a[m * Fa + f] : b[m * Fb + (f - Fa)]);
}

// ---------------------------------------------------------------------------
// Host-side launch helpers
// ---------------------------------------------------------------------------
static void launch_gemm(hipStream_t st, int nsub,
                        const _Float16* A, int lda, const _Float16* Bt,
                        const float* Cin, int ldcin,
                        const float* b1, const float* b2,
                        const float* bng, const float* bnb, int relu,
                        float* C, _Float16* Ch, int ldc,
                        int M, int Nreal, int Npad, int K) {
  int blocks = (M / 64) * (Npad / (16 * nsub));    // exact: no tail blocks
  size_t shmem = (size_t)16 * nsub * K * sizeof(_Float16);
  if (nsub == 4)
    wmma_gemm<4><<<blocks, 128, shmem, st>>>(A, lda, Bt, Cin, ldcin, b1, b2,
                                             bng, bnb, relu, C, Ch, ldc,
                                             M, Nreal, Npad, K);
  else
    wmma_gemm<1><<<blocks, 128, shmem, st>>>(A, lda, Bt, Cin, ldcin, b1, b2,
                                             bng, bnb, relu, C, Ch, ldc,
                                             M, Nreal, Npad, K);
}

static inline int cdiv(long long a, long long b) { return (int)((a + b - 1) / b); }

extern "C" void kernel_launch(void* const* d_in, const int* in_sizes, int n_in,
                              void* d_out, int out_size, void* d_ws, size_t ws_size,
                              hipStream_t stream) {
  const float* dyn    = (const float*)d_in[0];   // (N,T,16)
  const float* xstat  = (const float*)d_in[1];   // (N,128)
  const int*   eidx   = (const int*)d_in[2];     // (2,E)
  const float* gcn1_w = (const float*)d_in[3];   // (128,64)
  const float* gcn1_b = (const float*)d_in[4];
  const float* gcn2_w = (const float*)d_in[5];   // (64,64)
  const float* gcn2_b = (const float*)d_in[6];
  const float* bn1_g  = (const float*)d_in[7];
  const float* bn1_b  = (const float*)d_in[8];
  const float* bn2_g  = (const float*)d_in[9];
  const float* bn2_b  = (const float*)d_in[10];
  const float* w_ih0  = (const float*)d_in[11];  // (512,16)  (N,K) already
  const float* w_hh0  = (const float*)d_in[12];  // (512,128) (N,K) already
  const float* b_ih0  = (const float*)d_in[13];
  const float* b_hh0  = (const float*)d_in[14];
  const float* w_ih1  = (const float*)d_in[15];  // (512,128)
  const float* w_hh1  = (const float*)d_in[16];  // (512,128)
  const float* b_ih1  = (const float*)d_in[17];
  const float* b_hh1  = (const float*)d_in[18];
  const float* fc1_w  = (const float*)d_in[19];  // (192,128) (K,N)
  const float* fc1_b  = (const float*)d_in[20];
  const float* fbn_g  = (const float*)d_in[21];
  const float* fbn_b  = (const float*)d_in[22];
  const float* fc2_w  = (const float*)d_in[23];  // (128,64)  (K,N)
  const float* fc2_b  = (const float*)d_in[24];
  const float* out_w  = (const float*)d_in[25];  // (64,10)   (K,N)
  const float* out_b  = (const float*)d_in[26];

  const int* src = eidx;
  const int* dst = eidx + N_EDGES;

  // ---- workspace carve-up, everything 256B aligned ----
  float* ws = (float*)d_ws;
  size_t o = 0;
  auto af32 = [&](size_t n) { float* p = ws + o; o += (n + 63) & ~(size_t)63; return p; };
  auto af16 = [&](size_t n) { _Float16* p = (_Float16*)(ws + o);
                              o += ((n + 1) / 2 + 63) & ~(size_t)63; return p; };
  // fp32 buffers
  float* deg    = af32(N_NODES);
  float* dis    = af32(N_NODES);
  float* hpre   = af32((size_t)N_NODES * G_HID);     // pre-aggregation features
  float* agg    = af32((size_t)N_NODES * G_HID);     // scatter accumulator
  float* gfin   = af32((size_t)N_NODES * G_HID);     // finish f32 out (gnn emb on layer 2)
  float* h0     = af32((size_t)N_NODES * R_HID);     // LSTM states (h0,c0,h1,c1 contiguous)
  float* c0     = af32((size_t)N_NODES * R_HID);
  float* h1     = af32((size_t)N_NODES * R_HID);
  float* c1     = af32((size_t)N_NODES * R_HID);
  float* gates  = af32((size_t)N_NODES * 4 * R_HID);
  // f16 activation operands
  _Float16* xstat_h = af16((size_t)N_NODES * D_STAT);
  _Float16* dyn_h   = af16((size_t)N_NODES * T_STEPS * D_DYNP);  // K-padded 16->32
  _Float16* gout1_h = af16((size_t)N_NODES * G_HID);
  _Float16* h0_h    = af16((size_t)N_NODES * R_HID);
  _Float16* h1_h    = af16((size_t)N_NODES * R_HID);
  _Float16* fused_h = af16((size_t)N_NODES * (G_HID + R_HID));
  _Float16* f1_h    = af16((size_t)N_NODES * 128);
  _Float16* f2_h    = af16((size_t)N_NODES * 64);
  // f16 weights, stored (Npad, K) = B^T
  _Float16* gcn1_wh = af16((size_t)G_HID * D_STAT);
  _Float16* gcn2_wh = af16((size_t)G_HID * G_HID);
  _Float16* wih0_h  = af16((size_t)512 * D_DYNP);
  _Float16* whh0_h  = af16((size_t)512 * R_HID);
  _Float16* wih1_h  = af16((size_t)512 * R_HID);
  _Float16* whh1_h  = af16((size_t)512 * R_HID);
  _Float16* fc1_wh  = af16((size_t)128 * 192);
  _Float16* fc2_wh  = af16((size_t)64 * 128);
  _Float16* outw_h  = af16((size_t)N_OUTP * 64);
  (void)ws_size; (void)n_in; (void)in_sizes; (void)out_size;

  const int TB = 256;
  auto grid = [&](long long n) { return cdiv(n, TB); };

  // ---- operand staging: fp32 -> f16 (+transpose/pad) ----
  conv_padk_kernel<<<grid((long long)N_NODES * D_STAT), TB, 0, stream>>>(
      xstat, xstat_h, N_NODES, D_STAT, D_STAT);
  conv_padk_kernel<<<grid((long long)N_NODES * T_STEPS * D_DYNP), TB, 0, stream>>>(
      dyn, dyn_h, N_NODES * T_STEPS, D_DYN, D_DYNP);
  conv_padk_kernel<<<grid(512LL * D_DYNP), TB, 0, stream>>>(w_ih0, wih0_h, 512, D_DYN, D_DYNP);
  conv_padk_kernel<<<grid(512LL * R_HID), TB, 0, stream>>>(w_hh0, whh0_h, 512, R_HID, R_HID);
  conv_padk_kernel<<<grid(512LL * R_HID), TB, 0, stream>>>(w_ih1, wih1_h, 512, R_HID, R_HID);
  conv_padk_kernel<<<grid(512LL * R_HID), TB, 0, stream>>>(w_hh1, whh1_h, 512, R_HID, R_HID);
  conv_transpose_kernel<<<grid((long long)G_HID * D_STAT), TB, 0, stream>>>(
      gcn1_w, gcn1_wh, D_STAT, G_HID, G_HID);
  conv_transpose_kernel<<<grid((long long)G_HID * G_HID), TB, 0, stream>>>(
      gcn2_w, gcn2_wh, G_HID, G_HID, G_HID);
  conv_transpose_kernel<<<grid(128LL * 192), TB, 0, stream>>>(fc1_w, fc1_wh, 192, 128, 128);
  conv_transpose_kernel<<<grid(64LL * 128), TB, 0, stream>>>(fc2_w, fc2_wh, 128, 64, 64);
  conv_transpose_kernel<<<grid((long long)N_OUTP * 64), TB, 0, stream>>>(
      out_w, outw_h, 64, N_OUT, N_OUTP);

  // ---- degree / normalization (self loop -> deg starts at 1) ----
  fill_kernel<<<grid(N_NODES), TB, 0, stream>>>(deg, 1.0f, N_NODES);
  deg_accum_kernel<<<grid(N_EDGES), TB, 0, stream>>>(dst, deg, N_EDGES);
  dis_kernel<<<grid(N_NODES), TB, 0, stream>>>(deg, dis, N_NODES);

  // ---- GCN layer 1 ----
  launch_gemm(stream, 4, xstat_h, D_STAT, gcn1_wh, nullptr, 0,
              nullptr, nullptr, nullptr, nullptr, 0, hpre, nullptr, G_HID,
              N_NODES, G_HID, G_HID, D_STAT);
  fill_kernel<<<grid((long long)N_NODES * G_HID), TB, 0, stream>>>(agg, 0.0f, N_NODES * G_HID);
  gcn_scatter_kernel<<<grid((long long)N_EDGES * G_HID), TB, 0, stream>>>(
      hpre, src, dst, dis, agg, N_EDGES, G_HID);
  gcn_finish_kernel<<<grid((long long)N_NODES * G_HID), TB, 0, stream>>>(
      agg, hpre, dis, gcn1_b, bn1_g, bn1_b, gfin, gout1_h, N_NODES, G_HID);

  // ---- GCN layer 2 ----
  launch_gemm(stream, 4, gout1_h, G_HID, gcn2_wh, nullptr, 0,
              nullptr, nullptr, nullptr, nullptr, 0, hpre, nullptr, G_HID,
              N_NODES, G_HID, G_HID, G_HID);
  fill_kernel<<<grid((long long)N_NODES * G_HID), TB, 0, stream>>>(agg, 0.0f, N_NODES * G_HID);
  gcn_scatter_kernel<<<grid((long long)N_EDGES * G_HID), TB, 0, stream>>>(
      hpre, src, dst, dis, agg, N_EDGES, G_HID);
  gcn_finish_kernel<<<grid((long long)N_NODES * G_HID), TB, 0, stream>>>(
      agg, hpre, dis, gcn2_b, bn2_g, bn2_b, gfin, gout1_h, N_NODES, G_HID);
  // gfin now holds gnn_emb (f32)

  // ---- LSTM: zero h0,c0,h1,c1 (contiguous) ----
  fill_kernel<<<grid((long long)N_NODES * R_HID * 4), TB, 0, stream>>>(
      h0, 0.0f, N_NODES * R_HID * 4);

  const int G4 = 4 * R_HID;  // 512 gate width
  const int cellBlocks = grid((long long)N_NODES * R_HID);
  for (int t = 0; t < T_STEPS; ++t) {
    // layer 0: gates = x_t @ w_ih0^T + b_ih0 + b_hh0
    launch_gemm(stream, 4, dyn_h + (size_t)t * D_DYNP, T_STEPS * D_DYNP, wih0_h,
                nullptr, 0, b_ih0, b_hh0, nullptr, nullptr, 0,
                gates, nullptr, G4, N_NODES, G4, G4, D_DYNP);
    if (t > 0)  // gates += h0 @ w_hh0^T
      launch_gemm(stream, 4, h0_h, R_HID, whh0_h, gates, G4,
                  nullptr, nullptr, nullptr, nullptr, 0,
                  gates, nullptr, G4, N_NODES, G4, G4, R_HID);
    lstm_cell_kernel<<<cellBlocks, TB, 0, stream>>>(gates, G4, h0, h0_h, c0, N_NODES, R_HID);

    // layer 1: gates = h0_t @ w_ih1^T + b_ih1 + b_hh1
    launch_gemm(stream, 4, h0_h, R_HID, wih1_h, nullptr, 0,
                b_ih1, b_hh1, nullptr, nullptr, 0,
                gates, nullptr, G4, N_NODES, G4, G4, R_HID);
    if (t > 0)  // gates += h1 @ w_hh1^T
      launch_gemm(stream, 4, h1_h, R_HID, whh1_h, gates, G4,
                  nullptr, nullptr, nullptr, nullptr, 0,
                  gates, nullptr, G4, N_NODES, G4, G4, R_HID);
    lstm_cell_kernel<<<cellBlocks, TB, 0, stream>>>(gates, G4, h1, h1_h, c1, N_NODES, R_HID);
  }

  // ---- head: concat -> fc1(+BN+ReLU) -> fc2(+ReLU) -> out ----
  concat_kernel<<<grid((long long)N_NODES * (G_HID + R_HID)), TB, 0, stream>>>(
      gfin, G_HID, h1, R_HID, fused_h, N_NODES);
  launch_gemm(stream, 4, fused_h, G_HID + R_HID, fc1_wh, nullptr, 0,
              fc1_b, nullptr, fbn_g, fbn_b, 1,
              nullptr, f1_h, 128, N_NODES, 128, 128, G_HID + R_HID);
  launch_gemm(stream, 4, f1_h, 128, fc2_wh, nullptr, 0,
              fc2_b, nullptr, nullptr, nullptr, 1,
              nullptr, f2_h, 64, N_NODES, 64, 64, 128);
  launch_gemm(stream, 1, f2_h, 64, outw_h, nullptr, 0,
              out_b, nullptr, nullptr, nullptr, 0,
              (float*)d_out, nullptr, N_OUT, N_NODES, N_OUT, N_OUTP, 64);
}
